// CPNKIFluxDualAttention_54743653155408
// MI455X (gfx1250) — matrix operand order repaired and checked by reading, not compiled, not run on hardware
//
#include <hip/hip_runtime.h>
#include <hip/hip_bf16.h>

// ---------------------------------------------------------------------------
// Types for CDNA5 WMMA (wave32): v_wmma_f32_16x16x32_bf16
// ---------------------------------------------------------------------------
typedef __attribute__((ext_vector_type(16))) __bf16 v16bf;
typedef __attribute__((ext_vector_type(8)))  __bf16 v8bf;
typedef __attribute__((ext_vector_type(8)))  float  v8f;

#define WMMA_BF16(A, B, C) \
  __builtin_amdgcn_wmma_f32_16x16x32_bf16(false, (A), false, (B), (short)0, (C), false, false)

__device__ __forceinline__ v16bf frag_cat(v8bf lo, v8bf hi) {
  return __builtin_shufflevector(lo, hi, 0,1,2,3,4,5,6,7,8,9,10,11,12,13,14,15);
}
__device__ __forceinline__ v16bf ld_frag(const __bf16* p0, const __bf16* p1) {
  return frag_cat(*(const v8bf*)p0, *(const v8bf*)p1);
}

// Async global->LDS copy (CDNA5 GLOBAL_LOAD_ASYNC_TO_LDS_B128, ASYNCcnt).
// lds_off: byte offset into the workgroup LDS allocation (low 32 bits of the
// flat shared-pointer address per the LDS aperture rule).
__device__ __forceinline__ void async_ld_b128(unsigned lds_off, const void* gaddr) {
  asm volatile("global_load_async_to_lds_b128 %0, %1, off"
               :: "v"(lds_off), "v"(gaddr) : "memory");
}
__device__ __forceinline__ void wait_asynccnt0() {
  asm volatile("s_wait_asynccnt 0x0" ::: "memory");
}
__device__ __forceinline__ unsigned lds_offset_of(const void* shared_ptr) {
  return (unsigned)(unsigned long long)shared_ptr;
}

// Problem constants
#define S_IMG 2048
#define S_TXT 512
#define S_TOT 2560
#define NHEAD 24
#define HD    128
#define DIM   3072

// ---------------------------------------------------------------------------
// GEMM: C[M,N] = A[M,K] @ B[K,N] + bias   (A,B fp32 in memory, bf16 WMMA math)
// Block: 256 threads = 8 waves, tile 128x128, K-step 32, double-buffered LDS.
// ---------------------------------------------------------------------------
#define GLP 48  // LDS row stride (bf16 elems): 96B, 16B-aligned fragment chunks

__global__ __launch_bounds__(256) void gemm_bf16wmma(
    const float* __restrict__ A, const float* __restrict__ B,
    const float* __restrict__ bias, float* __restrict__ C,
    int M, int N, int K)
{
  __shared__ __bf16 As[2][128 * GLP];  // [m][k]
  __shared__ __bf16 Bs[2][128 * GLP];  // [n][k] (transposed stage)

  const int tid = threadIdx.x;
  const int lane = tid & 31, wave = tid >> 5;
  const int lane16 = lane & 15, half = lane >> 4;
  const int wm = wave >> 2;   // 0..1  -> 64-row strip
  const int wn = wave & 3;    // 0..3  -> 32-col strip
  const int mBase = blockIdx.y * 128;
  const int nBase = blockIdx.x * 128;

  v8f acc[4][2];
  for (int i = 0; i < 4; ++i)
    for (int j = 0; j < 2; ++j)
      acc[i][j] = (v8f){0.f,0.f,0.f,0.f,0.f,0.f,0.f,0.f};

  const int ar = tid >> 3;            // 0..31
  const int ac = (tid & 7) * 4;       // 0..28
  const int bk = tid >> 5;            // 0..7
  const int bn = (tid & 31) * 4;      // 0..124

  const float* Abase = A + (size_t)(mBase + ar) * K + ac;  // + rb*32*K + k0
  const float* Bbase = B + (size_t)bk * N + nBase + bn;    // + (k0+rb*8)*N

  // --- prologue: stage tile 0 into buffer 0
  for (int rb = 0; rb < 4; ++rb) {
    const float4 f = *(const float4*)(Abase + (size_t)rb * 32 * K);
    __bf16* d = &As[0][(rb * 32 + ar) * GLP + ac];
    d[0] = (__bf16)f.x; d[1] = (__bf16)f.y; d[2] = (__bf16)f.z; d[3] = (__bf16)f.w;
  }
  for (int rb = 0; rb < 4; ++rb) {
    const float4 f = *(const float4*)(Bbase + (size_t)rb * 8 * N);
    const int kk = rb * 8 + bk;
    Bs[0][(bn + 0) * GLP + kk] = (__bf16)f.x;
    Bs[0][(bn + 1) * GLP + kk] = (__bf16)f.y;
    Bs[0][(bn + 2) * GLP + kk] = (__bf16)f.z;
    Bs[0][(bn + 3) * GLP + kk] = (__bf16)f.w;
  }
  __syncthreads();

  int buf = 0;
  for (int k0 = 0; k0 < K; k0 += 32) {
    const bool more = (k0 + 32) < K;
    float4 fa[4], fb[4];
    if (more) {
      // global loads for next tile (land in regs while WMMAs run on LDS[buf])
      for (int rb = 0; rb < 4; ++rb)
        fa[rb] = *(const float4*)(Abase + (size_t)rb * 32 * K + (k0 + 32));
      for (int rb = 0; rb < 4; ++rb)
        fb[rb] = *(const float4*)(Bbase + (size_t)(k0 + 32 + rb * 8) * N);
      if (k0 + 64 < K) {
        __builtin_prefetch(Abase + (k0 + 64), 0, 3);
        __builtin_prefetch(Bbase + (size_t)(k0 + 64) * N, 0, 3);
      }
    }

    // --- compute on LDS[buf]
    v16bf af[4], bfr[2];
    for (int ms = 0; ms < 4; ++ms) {
      const int Mr = wm * 64 + ms * 16 + lane16;
      af[ms] = ld_frag(&As[buf][Mr * GLP + half * 8], &As[buf][Mr * GLP + 16 + half * 8]);
    }
    for (int ns = 0; ns < 2; ++ns) {
      const int Nc = wn * 32 + ns * 16 + lane16;
      bfr[ns] = ld_frag(&Bs[buf][Nc * GLP + half * 8], &Bs[buf][Nc * GLP + 16 + half * 8]);
    }
    for (int ms = 0; ms < 4; ++ms)
      for (int ns = 0; ns < 2; ++ns)
        acc[ms][ns] = WMMA_BF16(af[ms], bfr[ns], acc[ms][ns]);

    // --- write next tile into the other buffer
    if (more) {
      const int nb = buf ^ 1;
      for (int rb = 0; rb < 4; ++rb) {
        __bf16* d = &As[nb][(rb * 32 + ar) * GLP + ac];
        d[0] = (__bf16)fa[rb].x; d[1] = (__bf16)fa[rb].y;
        d[2] = (__bf16)fa[rb].z; d[3] = (__bf16)fa[rb].w;
      }
      for (int rb = 0; rb < 4; ++rb) {
        const int kk = rb * 8 + bk;
        Bs[nb][(bn + 0) * GLP + kk] = (__bf16)fb[rb].x;
        Bs[nb][(bn + 1) * GLP + kk] = (__bf16)fb[rb].y;
        Bs[nb][(bn + 2) * GLP + kk] = (__bf16)fb[rb].z;
        Bs[nb][(bn + 3) * GLP + kk] = (__bf16)fb[rb].w;
      }
    }
    __syncthreads();
    buf ^= 1;
  }

  // --- epilogue: bias always present, no branches
  for (int ms = 0; ms < 4; ++ms)
    for (int ns = 0; ns < 2; ++ns) {
      const int col = nBase + wn * 32 + ns * 16 + lane16;
      const float bv = bias[col];
      for (int v = 0; v < 8; ++v) {
        const int row = mBase + wm * 64 + ms * 16 + v + 8 * half;
        C[(size_t)row * N + col] = acc[ms][ns][v] + bv;
      }
    }
}

// ---------------------------------------------------------------------------
// Per-(token,head): RMSNorm -> RoPE -> bf16, into [head][2560][128]
// Wave-shuffle reductions; RoPE partner via shfl_xor(1).
// ---------------------------------------------------------------------------
__global__ __launch_bounds__(128) void qk_post(
    const float* __restrict__ X, const float* __restrict__ nw,
    const float* __restrict__ cosp, const float* __restrict__ sinp,
    __bf16* __restrict__ out, int seqOff, float scale)
{
  __shared__ float part[4];
  const int t = threadIdx.x, s = blockIdx.x, h = blockIdx.y;
  const int lane = t & 31, wave = t >> 5;
  const float x = X[(size_t)s * DIM + h * HD + t];

  float ss = x * x;
  for (int msk = 1; msk < 32; msk <<= 1) ss += __shfl_xor(ss, msk, 32);
  if (lane == 0) part[wave] = ss;
  __syncthreads();
  const float tot = part[0] + part[1] + part[2] + part[3];
  const float inv = rsqrtf(tot * (1.f / 128.f) + 1e-6f);

  const float xn = x * inv * nw[t];
  const float xp = __shfl_xor(xn, 1, 32);          // partner element t^1
  const float rot = (t & 1) ? xp : -xp;             // [-x1, x0] pairs
  const float y = (xn * cosp[(size_t)s * HD + t] + rot * sinp[(size_t)s * HD + t]) * scale;
  out[((size_t)h * S_TOT + seqOff + s) * HD + t] = (__bf16)y;
}

__global__ __launch_bounds__(128) void v_reorder(
    const float* __restrict__ V, __bf16* __restrict__ out, int seqOff)
{
  const int t = threadIdx.x, s = blockIdx.x, h = blockIdx.y;
  out[((size_t)h * S_TOT + seqOff + s) * HD + t] = (__bf16)V[(size_t)s * DIM + h * HD + t];
}

// ---------------------------------------------------------------------------
// Flash attention: grid (S_TOT/64, 24), 128 threads = 4 waves.
// Each wave owns 16 queries; K/V processed in 32-key blocks via LDS.
// K tile staged with GLOBAL_LOAD_ASYNC_TO_LDS_B128 (ASYNCcnt path).
// ---------------------------------------------------------------------------
#define KLP 144   // Ks row stride (bf16): [key][d]
#define VLP 48    // Vt row stride (bf16): [d][key]

__global__ __launch_bounds__(128) void flash_attn(
    const __bf16* __restrict__ Qa, const __bf16* __restrict__ Ka,
    const __bf16* __restrict__ Va, float* __restrict__ joint)
{
  __shared__ __bf16 Ks[32 * KLP];
  __shared__ __bf16 Vt[HD * VLP];
  __shared__ __bf16 Psc[4 * 16 * 32];

  const int tid = threadIdx.x;
  const int lane = tid & 31, wave = tid >> 5;
  const int lane16 = lane & 15, half = lane >> 4;
  const int h = blockIdx.y;
  const int qb = blockIdx.x * 64 + wave * 16;

  // Load Q fragments for this wave's 16 rows (4 A-frags covering d=0..127)
  v16bf qf[4];
  {
    const __bf16* qp = Qa + ((size_t)h * S_TOT + qb + lane16) * HD;
    for (int ds = 0; ds < 4; ++ds)
      qf[ds] = ld_frag(qp + ds * 32 + half * 8, qp + ds * 32 + 16 + half * 8);
  }

  v8f o[8];
  for (int i = 0; i < 8; ++i) o[i] = (v8f){0.f,0.f,0.f,0.f,0.f,0.f,0.f,0.f};
  float m_i[8], l_i[8];
  for (int v = 0; v < 8; ++v) { m_i[v] = -1e30f; l_i[v] = 0.f; }

  const int skk = tid & 31, sdg = tid >> 5;  // staging: key row, 32-d group
  __bf16* Pw = &Psc[wave * 16 * 32];
  const size_t kvBase = (size_t)h * S_TOT;
  const unsigned ksOff = lds_offset_of(&Ks[skk * KLP + sdg * 32]);

  for (int kb = 0; kb < S_TOT; kb += 32) {
    // --- stage K block [32 keys][128 d] via async global->LDS (no VGPR hop)
    {
      const __bf16* sk = Ka + (kvBase + kb + skk) * HD + sdg * 32;
      for (int j = 0; j < 4; ++j)
        async_ld_b128(ksOff + j * 16, sk + j * 8);
      // V transposed [128 d][32 keys] (manual: transpose during store)
      const __bf16* sv = Va + (kvBase + kb + skk) * HD + sdg * 32;
      for (int j = 0; j < 4; ++j) {
        const v8bf vv = *(const v8bf*)(sv + j * 8);
        for (int e = 0; e < 8; ++e) Vt[(sdg * 32 + j * 8 + e) * VLP + skk] = vv[e];
      }
      if (kb + 32 < S_TOT) {  // prefetch next block through GL2
        __builtin_prefetch(Ka + (kvBase + kb + 32 + skk) * HD + sdg * 32, 0, 3);
        __builtin_prefetch(Va + (kvBase + kb + 32 + skk) * HD + sdg * 32, 0, 3);
      }
    }
    wait_asynccnt0();
    __syncthreads();

    // --- scores: S[16 q][32 k] via 2 N-tiles, 4 d-steps each
    v8f s0 = (v8f){0.f,0.f,0.f,0.f,0.f,0.f,0.f,0.f};
    v8f s1 = s0;
    for (int ds = 0; ds < 4; ++ds) {
      const v16bf b0 = ld_frag(&Ks[lane16 * KLP + ds * 32 + half * 8],
                               &Ks[lane16 * KLP + ds * 32 + 16 + half * 8]);
      const v16bf b1 = ld_frag(&Ks[(16 + lane16) * KLP + ds * 32 + half * 8],
                               &Ks[(16 + lane16) * KLP + ds * 32 + 16 + half * 8]);
      s0 = WMMA_BF16(qf[ds], b0, s0);
      s1 = WMMA_BF16(qf[ds], b1, s1);
    }

    // --- online softmax (row stats per accumulator index v = M - 8*half)
    float alpha[8];
    for (int v = 0; v < 8; ++v) {
      float mv = fmaxf(s0[v], s1[v]);
      for (int msk = 1; msk < 16; msk <<= 1) mv = fmaxf(mv, __shfl_xor(mv, msk, 32));
      const float mnew = fmaxf(m_i[v], mv);
      alpha[v] = __expf(m_i[v] - mnew);
      const float p0 = __expf(s0[v] - mnew);
      const float p1 = __expf(s1[v] - mnew);
      float rs = p0 + p1;
      for (int msk = 1; msk < 16; msk <<= 1) rs += __shfl_xor(rs, msk, 32);
      l_i[v] = l_i[v] * alpha[v] + rs;
      m_i[v] = mnew;
      // write P tile row-major [16][32] into per-wave scratch
      Pw[(v + 8 * half) * 32 + lane16]      = (__bf16)p0;
      Pw[(v + 8 * half) * 32 + 16 + lane16] = (__bf16)p1;
    }
    for (int df = 0; df < 8; ++df)
      for (int v = 0; v < 8; ++v) o[df][v] *= alpha[v];
    __syncthreads();

    // --- P @ V: A = P (16x32 keys), B = Vt columns (d)
    const v16bf pf = ld_frag(&Pw[lane16 * 32 + half * 8],
                             &Pw[lane16 * 32 + 16 + half * 8]);
    for (int df = 0; df < 8; ++df) {
      const int dcol = df * 16 + lane16;
      const v16bf vf = ld_frag(&Vt[dcol * VLP + half * 8],
                               &Vt[dcol * VLP + 16 + half * 8]);
      o[df] = WMMA_BF16(pf, vf, o[df]);
    }
    __syncthreads();
  }

  // --- epilogue: normalize and scatter into joint [token][3072] fp32
  float invl[8];
  for (int v = 0; v < 8; ++v) invl[v] = 1.f / l_i[v];
  for (int df = 0; df < 8; ++df)
    for (int v = 0; v < 8; ++v) {
      const int tok = qb + v + 8 * half;
      joint[(size_t)tok * DIM + h * HD + df * 16 + lane16] = o[df][v] * invl[v];
    }
}

// ---------------------------------------------------------------------------
// Host: orchestrate the pipeline on `stream`
// ---------------------------------------------------------------------------
extern "C" void kernel_launch(void* const* d_in, const int* in_sizes, int n_in,
                              void* d_out, int out_size, void* d_ws, size_t ws_size,
                              hipStream_t stream) {
  (void)in_sizes; (void)n_in; (void)out_size; (void)ws_size;
  const float* hs      = (const float*)d_in[0];
  const float* enc     = (const float*)d_in[1];
  const float* img_cos = (const float*)d_in[2];
  const float* img_sin = (const float*)d_in[3];
  const float* txt_cos = (const float*)d_in[4];
  const float* txt_sin = (const float*)d_in[5];
  const float* wq  = (const float*)d_in[6];  const float* bq  = (const float*)d_in[7];
  const float* wk  = (const float*)d_in[8];  const float* bk  = (const float*)d_in[9];
  const float* wv  = (const float*)d_in[10]; const float* bv  = (const float*)d_in[11];
  const float* awq = (const float*)d_in[12]; const float* abq = (const float*)d_in[13];
  const float* awk = (const float*)d_in[14]; const float* abk = (const float*)d_in[15];
  const float* awv = (const float*)d_in[16]; const float* abv = (const float*)d_in[17];
  const float* wo  = (const float*)d_in[18]; const float* bo  = (const float*)d_in[19];
  const float* awo = (const float*)d_in[20]; const float* abo = (const float*)d_in[21];
  const float* nq  = (const float*)d_in[22]; const float* nk  = (const float*)d_in[23];
  const float* anq = (const float*)d_in[24]; const float* ank = (const float*)d_in[25];
  float* out = (float*)d_out;

  // Workspace carve-up
  char* ws = (char*)d_ws;
  size_t off = 0;
  auto take = [&](size_t bytes) -> char* {
    char* p = ws + off;
    off += (bytes + 255) & ~(size_t)255;
    return p;
  };
  float*  qimg = (float*)take((size_t)S_IMG * DIM * 4);
  float*  kimg = (float*)take((size_t)S_IMG * DIM * 4);
  float*  vimg = (float*)take((size_t)S_IMG * DIM * 4);
  float*  qtxt = (float*)take((size_t)S_TXT * DIM * 4);
  float*  ktxt = (float*)take((size_t)S_TXT * DIM * 4);
  float*  vtxt = (float*)take((size_t)S_TXT * DIM * 4);
  __bf16* Qa   = (__bf16*)take((size_t)NHEAD * S_TOT * HD * 2);
  __bf16* Ka   = (__bf16*)take((size_t)NHEAD * S_TOT * HD * 2);
  __bf16* Va   = (__bf16*)take((size_t)NHEAD * S_TOT * HD * 2);
  float*  joint= (float*)take((size_t)S_TOT * DIM * 4);

  const dim3 blkG(256), blkP(128);
  const dim3 grdImg(DIM / 128, S_IMG / 128);
  const dim3 grdTxt(DIM / 128, S_TXT / 128);

  // QKV projections (bf16 WMMA, fp32 accum, bias fused)
  gemm_bf16wmma<<<grdImg, blkG, 0, stream>>>(hs,  wq,  bq,  qimg, S_IMG, DIM, DIM);
  gemm_bf16wmma<<<grdImg, blkG, 0, stream>>>(hs,  wk,  bk,  kimg, S_IMG, DIM, DIM);
  gemm_bf16wmma<<<grdImg, blkG, 0, stream>>>(hs,  wv,  bv,  vimg, S_IMG, DIM, DIM);
  gemm_bf16wmma<<<grdTxt, blkG, 0, stream>>>(enc, awq, abq, qtxt, S_TXT, DIM, DIM);
  gemm_bf16wmma<<<grdTxt, blkG, 0, stream>>>(enc, awk, abk, ktxt, S_TXT, DIM, DIM);
  gemm_bf16wmma<<<grdTxt, blkG, 0, stream>>>(enc, awv, abv, vtxt, S_TXT, DIM, DIM);

  // RMSNorm + RoPE + head-major repack (txt at seq 0, img at seq 512)
  const float qscale = 0.08838834764831845f; // 1/sqrt(128)
  qk_post<<<dim3(S_IMG, NHEAD), blkP, 0, stream>>>(qimg, nq,  img_cos, img_sin, Qa, S_TXT, qscale);
  qk_post<<<dim3(S_IMG, NHEAD), blkP, 0, stream>>>(kimg, nk,  img_cos, img_sin, Ka, S_TXT, 1.0f);
  qk_post<<<dim3(S_TXT, NHEAD), blkP, 0, stream>>>(qtxt, anq, txt_cos, txt_sin, Qa, 0,     qscale);
  qk_post<<<dim3(S_TXT, NHEAD), blkP, 0, stream>>>(ktxt, ank, txt_cos, txt_sin, Ka, 0,     1.0f);
  v_reorder<<<dim3(S_IMG, NHEAD), blkP, 0, stream>>>(vimg, Va, S_TXT);
  v_reorder<<<dim3(S_TXT, NHEAD), blkP, 0, stream>>>(vtxt, Va, 0);

  // Joint flash attention -> joint [2560][3072] fp32
  flash_attn<<<dim3(S_TOT / 64, NHEAD), blkP, 0, stream>>>(Qa, Ka, Va, joint);

  // Output projections: img rows are joint[512:2560], txt rows joint[0:512]
  gemm_bf16wmma<<<grdImg, blkG, 0, stream>>>(joint + (size_t)S_TXT * DIM, wo, bo,
                                             out, S_IMG, DIM, DIM);
  gemm_bf16wmma<<<grdTxt, blkG, 0, stream>>>(joint, awo, abo,
                                             out + (size_t)S_IMG * DIM, S_TXT, DIM, DIM);
}
